// MQA_49572512530681
// MI455X (gfx1250) — matrix operand order repaired and checked by reading, not compiled, not run on hardware
//
#include <hip/hip_runtime.h>

// ---------------- problem constants (match reference) ----------------
constexpr int cB  = 4;
constexpr int cNQ = 2048;
constexpr int cNK = 2048;
constexpr int cD  = 1024;
constexpr int cH  = 8;
constexpr int cDK = 128;
constexpr int cDV = 128;
constexpr int cDO = 1024;

typedef __attribute__((ext_vector_type(16))) __bf16 v16bf;
typedef __attribute__((ext_vector_type(8)))  __bf16 v8bf;
typedef __attribute__((ext_vector_type(8)))  float  v8f;
typedef __attribute__((ext_vector_type(4)))  float  v4f;

__device__ __forceinline__ v16bf cat8(v8bf lo, v8bf hi) {
  return __builtin_shufflevector(lo, hi, 0, 1, 2, 3, 4, 5, 6, 7,
                                         8, 9, 10, 11, 12, 13, 14, 15);
}

// ---------------- WMMA fragment loaders (wave32, ISA 7.12.2 layouts) ----------------
// A 16x32 bf16: lanes 0-15 -> row M=lane,    K = {0..7, 16..23}
//               lanes 16-31 -> row M=lane-16, K = {8..15, 24..31}
__device__ __forceinline__ v16bf load_a_bf16(const __bf16* __restrict__ src, int ld, int lane) {
  const int m  = lane & 15;
  const int kb = (lane >> 4) << 3;                   // 0 or 8
  const __bf16* r = src + (size_t)m * ld + kb;
  v8bf lo = *(const v8bf*)(r);                       // K = kb .. kb+7
  v8bf hi = *(const v8bf*)(r + 16);                  // K = 16+kb .. 16+kb+7
  return cat8(lo, hi);
}

__device__ __forceinline__ v16bf load_a_f32(const float* __restrict__ src, int ld, int lane) {
  const int m  = lane & 15;
  const int kb = (lane >> 4) << 3;
  const float* r = src + (size_t)m * ld + kb;
  v4f f0 = *(const v4f*)(r);
  v4f f1 = *(const v4f*)(r + 4);
  v4f f2 = *(const v4f*)(r + 16);
  v4f f3 = *(const v4f*)(r + 20);
  v16bf a;
#pragma unroll
  for (int i = 0; i < 4; ++i) {
    a[i]      = (__bf16)f0[i];
    a[4 + i]  = (__bf16)f1[i];
    a[8 + i]  = (__bf16)f2[i];
    a[12 + i] = (__bf16)f3[i];
  }
  return a;
}

// B 32x16 where logical B[k][n] = W[n][k]  (row n contiguous along K)
// lanes 0-15 -> col N=lane, K={0..7,16..23} ; lanes 16-31 -> col N=lane-16, K={8..15,24..31}
__device__ __forceinline__ v16bf load_bt_bf16(const __bf16* __restrict__ src, long ld, int lane) {
  const int n  = lane & 15;
  const int kb = (lane >> 4) << 3;                   // 0 or 8
  const __bf16* r = src + (size_t)n * ld + kb;
  v8bf lo = *(const v8bf*)(r);
  v8bf hi = *(const v8bf*)(r + 16);
  return cat8(lo, hi);
}

__device__ __forceinline__ v16bf load_bt_f32(const float* __restrict__ src, int ld, int lane) {
  const int n  = lane & 15;
  const int kb = (lane >> 4) << 3;
  const float* r = src + (size_t)n * ld + kb;
  v4f f0 = *(const v4f*)(r);
  v4f f1 = *(const v4f*)(r + 4);
  v4f f2 = *(const v4f*)(r + 16);
  v4f f3 = *(const v4f*)(r + 20);
  v16bf b;
#pragma unroll
  for (int i = 0; i < 4; ++i) {
    b[i]      = (__bf16)f0[i];
    b[4 + i]  = (__bf16)f1[i];
    b[8 + i]  = (__bf16)f2[i];
    b[12 + i] = (__bf16)f3[i];
  }
  return b;
}

__device__ __forceinline__ v8f wmma_bf16(v16bf a, v16bf b, v8f c) {
  return __builtin_amdgcn_wmma_f32_16x16x32_bf16(false, a, false, b, (short)0, c, false, false);
}

// ---------------- Kernel 1: Y(M,N) = X(M,K) @ W(N,K)^T + bias -> bf16 ----------------
// TRANS=false: Y row-major (M,N).  TRANS=true: Y stored transposed (N,M) row-major.
// 32x32 wave tile (2x2 accums), block = 8 waves (4 along M, 2 along N) -> 128x64 tile.
// grid: (N/64, M/128)
template <bool TRANS>
__global__ __launch_bounds__(256) void k_proj(
    const float* __restrict__ X, const float* __restrict__ W,
    const float* __restrict__ bias, __bf16* __restrict__ Y, int N, int K, int M) {
  const int lane = threadIdx.x & 31;
  const int wave = threadIdx.x >> 5;
  const int row0 = blockIdx.y * 128 + (wave & 3) * 32;
  const int col0 = blockIdx.x * 64 + (wave >> 2) * 32;
  const float* X0 = X + (size_t)row0 * K;
  const float* X1 = X0 + (size_t)16 * K;
  const float* W0 = W + (size_t)col0 * K;
  const float* W1 = W0 + (size_t)16 * K;
  v8f c00 = {}, c01 = {}, c10 = {}, c11 = {};
  for (int k0 = 0; k0 < K; k0 += 32) {
    __builtin_prefetch(X0 + k0 + 64, 0, 0);
    v16bf a0 = load_a_f32(X0 + k0, K, lane);
    v16bf a1 = load_a_f32(X1 + k0, K, lane);
    v16bf b0 = load_bt_f32(W0 + k0, K, lane);
    v16bf b1 = load_bt_f32(W1 + k0, K, lane);
    c00 = wmma_bf16(a0, b0, c00);
    c01 = wmma_bf16(a0, b1, c01);
    c10 = wmma_bf16(a1, b0, c10);
    c11 = wmma_bf16(a1, b1, c11);
  }
  const int nn    = lane & 15;
  const int rbase = (lane >> 4) << 3;
#pragma unroll
  for (int tm = 0; tm < 2; ++tm) {
#pragma unroll
    for (int tn = 0; tn < 2; ++tn) {
      const v8f& c = tm ? (tn ? c11 : c10) : (tn ? c01 : c00);
      const int n  = col0 + tn * 16 + nn;
      const int r0 = row0 + tm * 16 + rbase;
      const float bv = bias[n];
      if (TRANS) {
        v8bf o;
#pragma unroll
        for (int i = 0; i < 8; ++i) o[i] = (__bf16)(c[i] + bv);
        *(v8bf*)(Y + (size_t)n * M + r0) = o;        // contiguous 16B store
      } else {
#pragma unroll
        for (int i = 0; i < 8; ++i)
          Y[(size_t)(r0 + i) * N + n] = (__bf16)(c[i] + bv);
      }
    }
  }
}

// ---------------- Kernel 2: S = qh @ kh^T * scale + mask -> fp32 (head-major) -----
// grid: (NK/64, NQ/128, B*H); 32x32 wave tile
__global__ __launch_bounds__(256) void k_scores(
    const __bf16* __restrict__ qh, const __bf16* __restrict__ kh,
    const float* __restrict__ mask, float* __restrict__ att) {
  const int lane = threadIdx.x & 31;
  const int wave = threadIdx.x >> 5;
  const int b = blockIdx.z / cH;
  const int h = blockIdx.z % cH;
  const int q0  = blockIdx.y * 128 + (wave & 3) * 32;
  const int kc0 = blockIdx.x * 64 + (wave >> 2) * 32;
  const int ldq = cH * cDK;
  const __bf16* Q0 = qh + (size_t)(b * cNQ + q0) * ldq + h * cDK;
  const __bf16* Q1 = Q0 + (size_t)16 * ldq;
  const __bf16* K0 = kh + (size_t)b * cNK * cDK + (size_t)kc0 * cDK;
  const __bf16* K1 = K0 + (size_t)16 * cDK;
  v8f c00 = {}, c01 = {}, c10 = {}, c11 = {};
#pragma unroll
  for (int k0 = 0; k0 < cDK; k0 += 32) {
    v16bf a0 = load_a_bf16(Q0 + k0, ldq, lane);
    v16bf a1 = load_a_bf16(Q1 + k0, ldq, lane);
    v16bf b0 = load_bt_bf16(K0 + k0, cDK, lane);
    v16bf b1 = load_bt_bf16(K1 + k0, cDK, lane);
    c00 = wmma_bf16(a0, b0, c00);
    c01 = wmma_bf16(a0, b1, c01);
    c10 = wmma_bf16(a1, b0, c10);
    c11 = wmma_bf16(a1, b1, c11);
  }
  float* outp       = att  + (size_t)(h * cB + b) * cNQ * cNK;
  const float* mrow = mask + (size_t)b * cNQ * cNK;
  const float scale = 0.08838834764831845f;          // 1/sqrt(128)
  const int nn    = lane & 15;
  const int rbase = (lane >> 4) << 3;
#pragma unroll
  for (int tm = 0; tm < 2; ++tm) {
#pragma unroll
    for (int tn = 0; tn < 2; ++tn) {
      const v8f& c = tm ? (tn ? c11 : c10) : (tn ? c01 : c00);
      const int n  = kc0 + tn * 16 + nn;
      const int r0 = q0 + tm * 16 + rbase;
#pragma unroll
      for (int i = 0; i < 8; ++i) {
        const size_t off = (size_t)(r0 + i) * cNK + n;
        outp[off] = c[i] * scale + mrow[off];
      }
    }
  }
}

// ---------------- Kernel 3: in-place row softmax over NK=2048 ----------------
__global__ __launch_bounds__(256) void k_softmax(float* __restrict__ att) {
  float* p = att + (size_t)blockIdx.x * cNK;
  const int t = threadIdx.x;
  float vals[8];
  float m = -3.402823466e38f;
#pragma unroll
  for (int i = 0; i < 8; ++i) { vals[i] = p[t + i * 256]; m = fmaxf(m, vals[i]); }
#pragma unroll
  for (int off = 16; off > 0; off >>= 1) m = fmaxf(m, __shfl_xor(m, off, 32));
  __shared__ float sm[8];
  __shared__ float ss[8];
  if ((t & 31) == 0) sm[t >> 5] = m;
  __syncthreads();
  float mm = sm[0];
#pragma unroll
  for (int j = 1; j < 8; ++j) mm = fmaxf(mm, sm[j]);
  float s = 0.f;
#pragma unroll
  for (int i = 0; i < 8; ++i) { vals[i] = __expf(vals[i] - mm); s += vals[i]; }
#pragma unroll
  for (int off = 16; off > 0; off >>= 1) s += __shfl_xor(s, off, 32);
  if ((t & 31) == 0) ss[t >> 5] = s;
  __syncthreads();
  float tot = 0.f;
#pragma unroll
  for (int j = 0; j < 8; ++j) tot += ss[j];
  const float inv = 1.0f / tot;
#pragma unroll
  for (int i = 0; i < 8; ++i) p[t + i * 256] = vals[i] * inv;
}

// ---------------- Kernel 4: outh = att @ vh -> bf16 (vh pre-transposed) ----------
// vhT layout: (DV, B*NK) row-major, so B[k][n] = vhT[n*(B*NK) + b*NK + k] (contig K)
// grid: (DV/64, NQ/128, B*H)
__global__ __launch_bounds__(256) void k_av(
    const float* __restrict__ att, const __bf16* __restrict__ vhT,
    __bf16* __restrict__ outh) {
  const int lane = threadIdx.x & 31;
  const int wave = threadIdx.x >> 5;
  const int b = blockIdx.z / cH;
  const int h = blockIdx.z % cH;
  const int q0 = blockIdx.y * 128 + (wave & 3) * 32;
  const int n0 = blockIdx.x * 64 + (wave >> 2) * 32;
  const long ldv = (long)cB * cNK;
  const float*  A0 = att + ((size_t)(h * cB + b) * cNQ + q0) * cNK;
  const float*  A1 = A0 + (size_t)16 * cNK;
  const __bf16* V0 = vhT + (size_t)n0 * ldv + (size_t)b * cNK;
  const __bf16* V1 = V0 + (size_t)16 * ldv;
  v8f c00 = {}, c01 = {}, c10 = {}, c11 = {};
  for (int k0 = 0; k0 < cNK; k0 += 32) {
    __builtin_prefetch(A0 + k0 + 64, 0, 0);
    v16bf a0 = load_a_f32(A0 + k0, cNK, lane);
    v16bf a1 = load_a_f32(A1 + k0, cNK, lane);
    v16bf b0 = load_bt_bf16(V0 + k0, ldv, lane);
    v16bf b1 = load_bt_bf16(V1 + k0, ldv, lane);
    c00 = wmma_bf16(a0, b0, c00);
    c01 = wmma_bf16(a0, b1, c01);
    c10 = wmma_bf16(a1, b0, c10);
    c11 = wmma_bf16(a1, b1, c11);
  }
  const int nn    = lane & 15;
  const int rbase = (lane >> 4) << 3;
#pragma unroll
  for (int tm = 0; tm < 2; ++tm) {
#pragma unroll
    for (int tn = 0; tn < 2; ++tn) {
      const v8f& c = tm ? (tn ? c11 : c10) : (tn ? c01 : c00);
      const int n  = n0 + tn * 16 + nn;
      const int r0 = q0 + tm * 16 + rbase;
#pragma unroll
      for (int i = 0; i < 8; ++i)
        outh[(size_t)(b * cNQ + r0 + i) * (cH * cDV) + h * cDV + n] = (__bf16)c[i];
    }
  }
}

// ---------------- Kernel 5: out = outh @ Wo^T + bo -> fp32 ----------------
// grid: (DO/64, (B*NQ)/128)
__global__ __launch_bounds__(256) void k_oproj(
    const __bf16* __restrict__ Xh, const float* __restrict__ Wo,
    const float* __restrict__ bo, float* __restrict__ out) {
  const int lane = threadIdx.x & 31;
  const int wave = threadIdx.x >> 5;
  const int row0 = blockIdx.y * 128 + (wave & 3) * 32;
  const int col0 = blockIdx.x * 64 + (wave >> 2) * 32;
  const int K = cH * cDV;                            // 1024
  const __bf16* X0 = Xh + (size_t)row0 * K;
  const __bf16* X1 = X0 + (size_t)16 * K;
  const float*  W0 = Wo + (size_t)col0 * K;
  const float*  W1 = W0 + (size_t)16 * K;
  v8f c00 = {}, c01 = {}, c10 = {}, c11 = {};
  for (int k0 = 0; k0 < K; k0 += 32) {
    __builtin_prefetch(W0 + k0 + 64, 0, 0);
    v16bf a0 = load_a_bf16(X0 + k0, K, lane);
    v16bf a1 = load_a_bf16(X1 + k0, K, lane);
    v16bf b0 = load_bt_f32(W0 + k0, K, lane);
    v16bf b1 = load_bt_f32(W1 + k0, K, lane);
    c00 = wmma_bf16(a0, b0, c00);
    c01 = wmma_bf16(a0, b1, c01);
    c10 = wmma_bf16(a1, b0, c10);
    c11 = wmma_bf16(a1, b1, c11);
  }
  const int nn    = lane & 15;
  const int rbase = (lane >> 4) << 3;
#pragma unroll
  for (int tm = 0; tm < 2; ++tm) {
#pragma unroll
    for (int tn = 0; tn < 2; ++tn) {
      const v8f& c = tm ? (tn ? c11 : c10) : (tn ? c01 : c00);
      const int n  = col0 + tn * 16 + nn;
      const int r0 = row0 + tm * 16 + rbase;
      const float bv = bo[n];
#pragma unroll
      for (int i = 0; i < 8; ++i)
        out[(size_t)(r0 + i) * cDO + n] = c[i] + bv;
    }
  }
}

// ---------------- host launcher ----------------
extern "C" void kernel_launch(void* const* d_in, const int* in_sizes, int n_in,
                              void* d_out, int out_size, void* d_ws, size_t ws_size,
                              hipStream_t stream) {
  (void)in_sizes; (void)n_in; (void)out_size; (void)ws_size;
  const float* q    = (const float*)d_in[0];
  const float* k    = (const float*)d_in[1];
  const float* v    = (const float*)d_in[2];
  const float* mask = (const float*)d_in[3];
  const float* Wq   = (const float*)d_in[4];
  const float* bq   = (const float*)d_in[5];
  const float* Wk   = (const float*)d_in[6];
  const float* bk   = (const float*)d_in[7];
  const float* Wv   = (const float*)d_in[8];
  const float* bv   = (const float*)d_in[9];
  const float* Wo   = (const float*)d_in[10];
  const float* bo   = (const float*)d_in[11];

  // workspace layout (bf16): qh | kh | vhT | outh  (~36 MB total)
  __bf16* qh   = (__bf16*)d_ws;
  __bf16* kh   = qh  + (size_t)cB * cNQ * cH * cDK;
  __bf16* vhT  = kh  + (size_t)cB * cNK * cDK;       // stored (DV, B*NK)
  __bf16* outh = vhT + (size_t)cB * cNK * cDV;

  float* att = (float*)d_out;                        // (H*B, NQ, NK)
  float* out = att + (size_t)cH * cB * cNQ * cNK;    // (B, NQ, DO)

  const dim3 blk(256);
  // projections (q,k row-major; v transposed for the AV GEMM)
  k_proj<false><<<dim3((cH * cDK) / 64, (cB * cNQ) / 128), blk, 0, stream>>>(
      q, Wq, bq, qh, cH * cDK, cD, cB * cNQ);
  k_proj<false><<<dim3(cDK / 64, (cB * cNK) / 128), blk, 0, stream>>>(
      k, Wk, bk, kh, cDK, cD, cB * cNK);
  k_proj<true><<<dim3(cDV / 64, (cB * cNK) / 128), blk, 0, stream>>>(
      v, Wv, bv, vhT, cDV, cD, cB * cNK);
  // scores + mask -> output attention region (head-major)
  k_scores<<<dim3(cNK / 64, cNQ / 128, cB * cH), blk, 0, stream>>>(qh, kh, mask, att);
  // in-place softmax
  k_softmax<<<dim3(cH * cB * cNQ), blk, 0, stream>>>(att);
  // att @ V
  k_av<<<dim3(cDV / 64, cNQ / 128, cB * cH), blk, 0, stream>>>(att, vhT, outh);
  // output projection
  k_oproj<<<dim3(cDO / 64, (cB * cNQ) / 128), blk, 0, stream>>>(outh, Wo, bo, out);
}
